// Graph_Attention_59957743452555
// MI455X (gfx1250) — matrix operand order repaired and checked by reading, not compiled, not run on hardware
//
#include <hip/hip_runtime.h>
#include <hip/hip_bf16.h>

// ---------------------------------------------------------------------------
// GAT forward for MI455X (gfx1250, wave32).
//   out = relu(scatter_add(x)[row] @ W + bias)        (linearity fusion!)
//   alpha = softmax(leaky_relu(out[row]@a_src + out[col]@a_dst))
//   result = out * alpha[:,None]
// GEMM uses v_wmma_f32_16x16x32_bf16 (bf16 A/B, f32 accum) — the only WMMA
// precision that clears the 23.3 TB/s bandwidth roofline (AI = 32 FLOP/B).
// W is staged in LDS pre-swizzled to fragment-major order so every B fragment
// is two ds_load_b128's straight into the WMMA source VGPRs (no reg shuffles).
// ---------------------------------------------------------------------------

typedef __attribute__((ext_vector_type(16))) __bf16 v16bf;
typedef __attribute__((ext_vector_type(8)))  float  v8f;

#define IN_F 128

// Pack two f32 -> one dword of 2 x bf16 (lo = a, hi = b) in 3 VALU ops:
// round-to-nearest via +0x8000 on the bit pattern, then a single v_perm_b32
// grabs both high halves: dst = {b[31:16], a[31:16]}.
__device__ __forceinline__ unsigned int pack2bf(float a, float b) {
    unsigned int ua = __float_as_uint(a) + 0x8000u;
    unsigned int ub = __float_as_uint(b) + 0x8000u;
    return __builtin_amdgcn_perm(ub, ua, 0x07060302u);  // v_perm_b32
}

// ---------------------------------------------------------------- zero d_out
__global__ __launch_bounds__(256) void k_zero(float4* out, int n4) {
    int i = blockIdx.x * 256 + threadIdx.x;
    if (i < n4) out[i] = make_float4(0.f, 0.f, 0.f, 0.f);
}

// ------------------------------------------------- scatter: acc[row] += x[col]
// one wave per edge; lane l owns 4 contiguous floats (float4 gather, 4 atomics)
__global__ __launch_bounds__(256) void k_scatter(const float* __restrict__ x,
                                                 const int* __restrict__ ei,
                                                 float* out, int E) {
    int gid = blockIdx.x * 256 + threadIdx.x;
    int e = gid >> 5, l = gid & 31;
    if (e >= E) return;
    int row = ei[e];
    int col = ei[E + e];
    float4 v = *(const float4*)(x + col * IN_F + l * 4);
    float* dst = out + row * IN_F + l * 4;
    unsafeAtomicAdd(dst + 0, v.x);   // global_atomic_add_f32
    unsafeAtomicAdd(dst + 1, v.y);
    unsafeAtomicAdd(dst + 2, v.z);
    unsafeAtomicAdd(dst + 3, v.w);
}

// ------------------------------------- in-place GEMM: out = relu(out@W + bias)
// 256 threads = 8 waves; wave computes a 16x128 tile with 32 bf16 WMMAs.
// LDS holds W as bf16 k-pairs in fragment-major order:
//   dword index = (((kb*2+g)*8 + nb)*16 + nl)*8 + v
// so lane (g,nl) reads fragment (kb,nb) as 8 contiguous dwords (2 x b128).
__global__ __launch_bounds__(256) void k_gemm_relu(float* out,
                                                   const float* __restrict__ W,
                                                   const float* __restrict__ bias,
                                                   int nrows) {
    __shared__ unsigned int wfr[4 * 2 * 8 * 16 * 8];   // 8192 dwords = 32 KB
    for (int idx = threadIdx.x; idx < 64 * 128; idx += 256) {
        int kp = idx >> 7, n = idx & 127;              // kp = K/2 (k-pair)
        int kb = kp >> 4, kpr = kp & 15;
        int gg = kpr >> 3, v = kpr & 7;                // K = 32kb + 16g + 2v
        int nb = n >> 4, nl2 = n & 15;
        wfr[((((kb * 2 + gg) * 8 + nb) * 16 + nl2) << 3) + v] =
            pack2bf(W[(2 * kp) * IN_F + n], W[(2 * kp + 1) * IN_F + n]);
    }
    __syncthreads();

    int wave = threadIdx.x >> 5;
    int lane = threadIdx.x & 31;
    int g  = lane >> 4;      // lane half: selects K half 16g..16g+15 (contiguous order)
    int nl = lane & 15;
    int m  = lane & 15;
    int row0 = (blockIdx.x * 8 + wave) * 16;
    if (row0 >= nrows) return;

    v8f c[8];
#pragma unroll
    for (int i = 0; i < 8; ++i)
#pragma unroll
        for (int j = 0; j < 8; ++j) c[i][j] = 0.f;

    const float* ap = out + (row0 + m) * IN_F;
#pragma unroll
    for (int kb = 0; kb < 4; ++kb) {
        // A fragment (16x32 bf16): lane(g,m) holds K = 16g+{0..15}
        const float4* p = (const float4*)(ap + kb * 32 + 16 * g);
        float4 f0 = p[0], f1 = p[1];
        const float4* q = (const float4*)(ap + kb * 32 + 8 + 16 * g);
        float4 f2 = q[0], f3 = q[1];
        union { v16bf v; unsigned int u[8]; } A;
        A.u[0] = pack2bf(f0.x, f0.y);
        A.u[1] = pack2bf(f0.z, f0.w);
        A.u[2] = pack2bf(f1.x, f1.y);
        A.u[3] = pack2bf(f1.z, f1.w);
        A.u[4] = pack2bf(f2.x, f2.y);
        A.u[5] = pack2bf(f2.z, f2.w);
        A.u[6] = pack2bf(f3.x, f3.y);
        A.u[7] = pack2bf(f3.z, f3.w);

        const uint4* fp0 =
            (const uint4*)&wfr[(((kb * 2 + g) * 8) * 16 + nl) << 3];
#pragma unroll
        for (int nb = 0; nb < 8; ++nb) {
            // B fragment (32x16 bf16): 8 contiguous dwords per lane in LDS
            union { v16bf v; uint4 qd[2]; } B;
            B.qd[0] = fp0[nb * 32 + 0];    // ds_load_b128
            B.qd[1] = fp0[nb * 32 + 1];    // ds_load_b128
            c[nb] = __builtin_amdgcn_wmma_f32_16x16x32_bf16(
                false, A.v, false, B.v, (short)0, c[nb], false, false);
        }
    }

    // store: C VGPR r -> row = row0 + 8g + r, col = nb*16 + nl ; fuse bias+relu
#pragma unroll
    for (int nb = 0; nb < 8; ++nb) {
        int col = nb * 16 + nl;
        float b = bias[col];
        float* op = out + (row0 + 8 * g) * IN_F + col;
#pragma unroll
        for (int r = 0; r < 8; ++r) {
            float v = c[nb][r] + b;
            op[r * IN_F] = v > 0.f ? v : 0.f;
        }
    }
}

// ------------------- per-edge logits: leaky_relu(out[row].a_src + out[col].a_dst)
__global__ __launch_bounds__(256) void k_logits(const float* __restrict__ out,
                                                const int* __restrict__ ei,
                                                const float* __restrict__ att,
                                                float* alpha, float* maxPart, int E) {
    __shared__ float smax[8];
    int gid = blockIdx.x * 256 + threadIdx.x;
    int e = gid >> 5, l = gid & 31;
    int wave = threadIdx.x >> 5;
    float logit = -3.4e38f;
    if (e < E) {
        int row = ei[e], col = ei[E + e];
        float4 orv = *(const float4*)(out + row * IN_F + l * 4);
        float4 ocv = *(const float4*)(out + col * IN_F + l * 4);
        float4 as  = *(const float4*)(att + l * 4);
        float4 ad  = *(const float4*)(att + IN_F + l * 4);
        float p = orv.x * as.x + orv.y * as.y + orv.z * as.z + orv.w * as.w
                + ocv.x * ad.x + ocv.y * ad.y + ocv.z * ad.z + ocv.w * ad.w;
#pragma unroll
        for (int off = 16; off > 0; off >>= 1) p += __shfl_xor(p, off, 32);
        if (l == 0) {
            p = p > 0.f ? p : 0.2f * p;   // leaky_relu(0.2)
            alpha[e] = p;
            logit = p;
        }
    }
    if (l == 0) smax[wave] = logit;
    __syncthreads();
    if (threadIdx.x == 0) {
        float mx = smax[0];
        for (int i = 1; i < 8; ++i) mx = fmaxf(mx, smax[i]);
        maxPart[blockIdx.x] = mx;
    }
}

// ---------------------------------------------------------- single-block reduces
__global__ __launch_bounds__(256) void k_reduce_max(const float* part, int n, float* res) {
    __shared__ float s[8];
    float m = -3.4e38f;
    for (int i = threadIdx.x; i < n; i += 256) m = fmaxf(m, part[i]);
#pragma unroll
    for (int off = 16; off > 0; off >>= 1) m = fmaxf(m, __shfl_xor(m, off, 32));
    if ((threadIdx.x & 31) == 0) s[threadIdx.x >> 5] = m;
    __syncthreads();
    if (threadIdx.x == 0) {
        float r = s[0];
        for (int i = 1; i < 8; ++i) r = fmaxf(r, s[i]);
        *res = r;
    }
}

__global__ __launch_bounds__(256) void k_reduce_sum(const float* part, int n, float* res) {
    __shared__ float s[8];
    float m = 0.f;
    for (int i = threadIdx.x; i < n; i += 256) m += part[i];
#pragma unroll
    for (int off = 16; off > 0; off >>= 1) m += __shfl_xor(m, off, 32);
    if ((threadIdx.x & 31) == 0) s[threadIdx.x >> 5] = m;
    __syncthreads();
    if (threadIdx.x == 0) {
        float r = 0.f;
        for (int i = 0; i < 8; ++i) r += s[i];
        *res = r;
    }
}

// ------------------------------------------- exp(alpha - gmax), partial sums
__global__ __launch_bounds__(256) void k_exp(float* alpha, const float* gmax,
                                             float* sumPart, int E) {
    __shared__ float s[8];
    int i = blockIdx.x * 256 + threadIdx.x;
    float mx = *gmax;
    float v = 0.f;
    if (i < E) {
        v = __expf(alpha[i] - mx);
        alpha[i] = v;
    }
#pragma unroll
    for (int off = 16; off > 0; off >>= 1) v += __shfl_xor(v, off, 32);
    if ((threadIdx.x & 31) == 0) s[threadIdx.x >> 5] = v;
    __syncthreads();
    if (threadIdx.x == 0) {
        float r = 0.f;
        for (int j = 0; j < 8; ++j) r += s[j];
        sumPart[blockIdx.x] = r;
    }
}

// ------------------------------------------------ final: out[i,:] *= alpha[i]/sum
__global__ __launch_bounds__(256) void k_scale(float* out, const float* __restrict__ alpha,
                                               const float* gsum, int N) {
    int gid = blockIdx.x * 256 + threadIdx.x;
    int i = gid >> 5, l = gid & 31;
    if (i >= N) return;
    float s = alpha[i] / (*gsum);
    float4* p = (float4*)(out + i * IN_F + l * 4);
    float4 v = *p;
    v.x *= s; v.y *= s; v.z *= s; v.w *= s;
    *p = v;
}

extern "C" void kernel_launch(void* const* d_in, const int* in_sizes, int n_in,
                              void* d_out, int out_size, void* d_ws, size_t ws_size,
                              hipStream_t stream) {
    const float* x    = (const float*)d_in[0];
    const int*   ei   = (const int*)  d_in[1];   // int32 (JAX x64 disabled)
    const float* W    = (const float*)d_in[2];
    const float* att  = (const float*)d_in[3];
    const float* bias = (const float*)d_in[4];
    float* out = (float*)d_out;

    int N = in_sizes[0] / IN_F;
    int E = in_sizes[1] / 2;

    float* ws       = (float*)d_ws;
    float* gmax     = ws + 0;
    float* gsum     = ws + 1;
    float* maxPart  = ws + 2;
    float* sumPart  = ws + 2 + 65536;
    float* alpha    = ws + 2 + 131072;

    int n4 = N * (IN_F / 4);
    k_zero<<<(n4 + 255) / 256, 256, 0, stream>>>((float4*)out, n4);

    int edgeBlocks = (E * 32 + 255) / 256;
    k_scatter<<<edgeBlocks, 256, 0, stream>>>(x, ei, out, E);

    int tileWaves = (N + 15) / 16;
    k_gemm_relu<<<(tileWaves + 7) / 8, 256, 0, stream>>>(out, W, bias, N);

    k_logits<<<edgeBlocks, 256, 0, stream>>>(out, ei, att, alpha, maxPart, E);
    k_reduce_max<<<1, 256, 0, stream>>>(maxPart, edgeBlocks, gmax);

    int eBlocks = (E + 255) / 256;
    k_exp<<<eBlocks, 256, 0, stream>>>(alpha, gmax, sumPart, E);
    k_reduce_sum<<<1, 256, 0, stream>>>(sumPart, eBlocks, gsum);

    k_scale<<<(N * 32 + 255) / 256, 256, 0, stream>>>(out, alpha, gsum, N);
}
